// D_MPNNLayer_69784628625694
// MI455X (gfx1250) — matrix operand order repaired and checked by reading, not compiled, not run on hardware
//
#include <hip/hip_runtime.h>
#include <stdint.h>

// ---------------------------------------------------------------------------
// MPNN layer on MI455X (gfx1250, wave32):
//   GEMM1 (E x 384 @ 384 x 128) -> relu -> atomic segment-sum -> bf16 cast
//   GEMM2 (E x 128 @ 128 x 128) -> snorm scale -> LayerNorm -> relu
// Matrix math: v_wmma_f32_16x16x32_bf16 (f32 accumulate).
// Weights staged into LDS per block via async global->LDS (ASYNCcnt path).
// B-fragment LDS layout is half-fragment interleaved so each ds_load_b128
// is bank-conflict-free (lane addresses stride 16B over a 256B bank row).
// Workspace layout (bytes):
//   [0,            20,480,000)  m_node f32   [40000][128]
//   [20,480,000,   30,720,000)  m_node bf16  [40000][128]
//   [30,720,000,   40,960,000)  x bf16       [40000][128]
//   [40,960,000,   41,058,304)  W1 B-fragments bf16 (96 frags x 512)
//   [41,058,304,   41,091,072)  W2 B-fragments bf16 (32 frags x 512)
// ---------------------------------------------------------------------------

#define DD 128

typedef __bf16 bf16_t;
typedef bf16_t v16bf __attribute__((ext_vector_type(16)));
typedef float  v8f   __attribute__((ext_vector_type(8)));
typedef int    v4i_t __attribute__((ext_vector_type(4)));

union AFrag { v16bf v; uint16_t h[16]; uint4 q[2]; };
union BFrag { v16bf v; uint4 q[2]; };

#if __has_builtin(__builtin_amdgcn_global_load_async_to_lds_b128) && \
    __has_builtin(__builtin_amdgcn_s_wait_asynccnt)
#define HAVE_ASYNC_LDS 1
#else
#define HAVE_ASYNC_LDS 0
#endif

__device__ __forceinline__ uint16_t f2bf(float f) {
  // round-to-nearest-even f32 -> bf16 bits (used only in small prep kernels)
  uint32_t u = __float_as_uint(f);
  u += 0x7FFFu + ((u >> 16) & 1u);
  return (uint16_t)(u >> 16);
}

// Stage n uint4's of global data into LDS, whole block cooperates.
__device__ __forceinline__ void stage_to_lds(uint4* lds, const uint4* g, int n) {
  const int tid = threadIdx.x;
  const int nt  = blockDim.x;
#if HAVE_ASYNC_LDS
  for (int i = tid; i < n; i += nt) {
    __builtin_amdgcn_global_load_async_to_lds_b128(
        (v4i_t*)(g + i), (v4i_t*)(lds + i), 0, 0);
  }
  __builtin_amdgcn_s_wait_asynccnt(0);
  __syncthreads();
#else
  for (int i = tid; i < n; i += nt) lds[i] = g[i];
  __syncthreads();
#endif
}

// ---------------- zero the node mailbox -----------------------------------
__global__ void zero_f32_kernel(float4* __restrict__ p, int n4) {
  int i = blockIdx.x * blockDim.x + threadIdx.x;
  if (i < n4) p[i] = float4{0.f, 0.f, 0.f, 0.f};
}

// ---------------- pack W (row-major [128][ld]) into WMMA B-fragments.
// Storage (bf16 units): off = frag*512 + halfq*256 + lane*8 + w,
// where frag = nb*KB+kb, t = halfq*8+w, column n = lane&15,
// K = halfq*16 + (lane>=16 ? 8 : 0) + w  (VGPR j holds t=2j,2j+1 packed lo/hi).
__global__ void pack_wB_kernel(const float* __restrict__ W, uint16_t* __restrict__ out,
                               int KB, int ld, int total) {
  int idx = blockIdx.x * blockDim.x + threadIdx.x;
  if (idx >= total) return;
  int w     = idx & 7;
  int lane  = (idx >> 3) & 31;
  int halfq = (idx >> 8) & 1;
  int f     = idx >> 9;
  int kb    = f % KB;
  int nb    = f / KB;
  int t     = halfq * 8 + w;
  int n     = lane & 15;
  int k     = (halfq << 4) + ((lane >> 4) << 3) + w;
  out[idx] = f2bf(W[(size_t)(nb * 16 + n) * ld + kb * 32 + k]);
}

// ---------------- f32 -> bf16 (packed 2 per dword), used for x and m_node --
__global__ void cvt_bf16_kernel(const float* __restrict__ in, uint32_t* __restrict__ out, int n2) {
  int i = blockIdx.x * blockDim.x + threadIdx.x;
  if (i < n2) {
    uint32_t lo = f2bf(in[2 * i]);
    uint32_t hi = f2bf(in[2 * i + 1]);
    out[i] = lo | (hi << 16);
  }
}

// Read one B fragment from LDS (conflict-free: lane addr stride = 16B).
__device__ __forceinline__ void load_bfrag(BFrag& b, const uint4* lds, int frag, int lane) {
  const uint4* p = lds + frag * 64 + lane;
  b.q[0] = p[0];
  b.q[1] = p[32];
}

// ---------------- GEMM1: m_edge = relu(concat(x[dst],x[src],h) @ W1^T),
// ---------------- scatter-add into m_node[dst].  One wave = 16 edges x 128 out.
__global__ __launch_bounds__(256) void gemm1_scatter_kernel(
    const uint16_t* __restrict__ xbf, const float* __restrict__ h,
    const uint16_t* __restrict__ w1b,
    const int* __restrict__ src, const int* __restrict__ dst,
    float* __restrict__ mnode) {
  extern __shared__ uint4 w1s[];          // 6144 uint4 = 96 KB (of 320 KB/WGP)
  stage_to_lds(w1s, (const uint4*)w1b, 6144);

  const int wave = (blockIdx.x * blockDim.x + threadIdx.x) >> 5;
  const int lane = threadIdx.x & 31;
  const int lo   = lane & 15;
  const int hi   = lane >> 4;
  const int e0   = wave * 16;

  // 16-bit A layout: both lane halves read the same 16 rows, K halves differ.
  const int eRow = e0 + lo;
  const int rDst = dst[eRow];
  const int rSrc = src[eRow];
  const uint16_t* xr0 = xbf + (size_t)rDst * DD;
  const uint16_t* xr1 = xbf + (size_t)rSrc * DD;
  const float*    hr  = h + (size_t)eRow * DD;

  const v8f vzero = {0.f, 0.f, 0.f, 0.f, 0.f, 0.f, 0.f, 0.f};
  v8f acc[8];
#pragma unroll
  for (int nb = 0; nb < 8; ++nb) acc[nb] = vzero;

#pragma unroll
  for (int kb = 0; kb < 12; ++kb) {
    AFrag a;
    if (kb < 8) {
      // bf16 source rows (pre-converted x): two b128 loads
      const uint16_t* r = (kb < 4) ? (xr0 + kb * 32) : (xr1 + (kb - 4) * 32);
      const uint4* ap = (const uint4*)(r + hi * 8);
      a.q[0] = ap[0];
      a.q[1] = ap[2];  // +16 bf16 elements
    } else {
      // f32 h rows: load + native cvt to bf16
      const float4* P4 = (const float4*)(hr + (kb - 8) * 32 + hi * 8);
      float4 f0 = P4[0], f1 = P4[1], f2 = P4[4], f3 = P4[5];
      a.v[0]  = (bf16_t)f0.x; a.v[1]  = (bf16_t)f0.y; a.v[2]  = (bf16_t)f0.z; a.v[3]  = (bf16_t)f0.w;
      a.v[4]  = (bf16_t)f1.x; a.v[5]  = (bf16_t)f1.y; a.v[6]  = (bf16_t)f1.z; a.v[7]  = (bf16_t)f1.w;
      a.v[8]  = (bf16_t)f2.x; a.v[9]  = (bf16_t)f2.y; a.v[10] = (bf16_t)f2.z; a.v[11] = (bf16_t)f2.w;
      a.v[12] = (bf16_t)f3.x; a.v[13] = (bf16_t)f3.y; a.v[14] = (bf16_t)f3.z; a.v[15] = (bf16_t)f3.w;
    }
    // batch B-fragment loads 4 at a time so ds_load latency overlaps WMMAs
#pragma unroll
    for (int g = 0; g < 2; ++g) {
      BFrag b[4];
#pragma unroll
      for (int j = 0; j < 4; ++j) load_bfrag(b[j], w1s, (g * 4 + j) * 12 + kb, lane);
#pragma unroll
      for (int j = 0; j < 4; ++j)
        acc[g * 4 + j] = __builtin_amdgcn_wmma_f32_16x16x32_bf16(
            false, a.v, false, b[j].v, (short)0, acc[g * 4 + j], false, false);
    }
  }

  // C/D layout: VGPR r, lane<16 -> (M=r, N=lo); lane>=16 -> (M=r+8, N=lo).
  int dIdx[8];
#pragma unroll
  for (int r = 0; r < 8; ++r) dIdx[r] = dst[e0 + hi * 8 + r];
#pragma unroll
  for (int nb = 0; nb < 8; ++nb) {
#pragma unroll
    for (int r = 0; r < 8; ++r) {
      float v = fmaxf(acc[nb][r], 0.f);
      unsafeAtomicAdd(&mnode[(size_t)dIdx[r] * DD + nb * 16 + lo], v);
    }
  }
}

// ---------------- GEMM2 + snorm + LayerNorm + relu -------------------------
__global__ __launch_bounds__(256) void gemm2_ln_kernel(
    const uint16_t* __restrict__ mbf, const uint16_t* __restrict__ w2b,
    const int* __restrict__ src, const float* __restrict__ snorm,
    const float* __restrict__ gamma, const float* __restrict__ beta,
    float* __restrict__ out) {
  extern __shared__ uint4 w2s[];          // 2048 uint4 = 32 KB
  stage_to_lds(w2s, (const uint4*)w2b, 2048);

  const int wave = (blockIdx.x * blockDim.x + threadIdx.x) >> 5;
  const int lane = threadIdx.x & 31;
  const int lo   = lane & 15;
  const int hi   = lane >> 4;
  const int e0   = wave * 16;

  const int srow = src[e0 + lo];
  const uint16_t* arow = mbf + (size_t)srow * DD;

  const v8f vzero = {0.f, 0.f, 0.f, 0.f, 0.f, 0.f, 0.f, 0.f};
  v8f acc[8];
#pragma unroll
  for (int nb = 0; nb < 8; ++nb) acc[nb] = vzero;

#pragma unroll
  for (int kb = 0; kb < 4; ++kb) {
    AFrag a;
    const uint4* ap = (const uint4*)(arow + kb * 32 + hi * 8);
    a.q[0] = ap[0];
    a.q[1] = ap[2];  // +16 bf16 elements
#pragma unroll
    for (int g = 0; g < 2; ++g) {
      BFrag b[4];
#pragma unroll
      for (int j = 0; j < 4; ++j) load_bfrag(b[j], w2s, (g * 4 + j) * 4 + kb, lane);
#pragma unroll
      for (int j = 0; j < 4; ++j)
        acc[g * 4 + j] = __builtin_amdgcn_wmma_f32_16x16x32_bf16(
            false, a.v, false, b[j].v, (short)0, acc[g * 4 + j], false, false);
    }
  }

  // per-edge scale (before LN, matches reference eps semantics)
  float s[8];
#pragma unroll
  for (int r = 0; r < 8; ++r) s[r] = snorm[e0 + hi * 8 + r];

  float sum[8], ssq[8];
#pragma unroll
  for (int r = 0; r < 8; ++r) { sum[r] = 0.f; ssq[r] = 0.f; }
#pragma unroll
  for (int nb = 0; nb < 8; ++nb) {
#pragma unroll
    for (int r = 0; r < 8; ++r) {
      float v = acc[nb][r] * s[r];
      acc[nb][r] = v;
      sum[r] += v;
      ssq[r] += v * v;
    }
  }
  // reduce across the 16 lanes of each half-wave (row m lives in one half)
#pragma unroll
  for (int mask = 1; mask < 16; mask <<= 1) {
#pragma unroll
    for (int r = 0; r < 8; ++r) {
      sum[r] += __shfl_xor(sum[r], mask, 32);
      ssq[r] += __shfl_xor(ssq[r], mask, 32);
    }
  }
  float mu[8], rstd[8];
#pragma unroll
  for (int r = 0; r < 8; ++r) {
    mu[r] = sum[r] * (1.f / 128.f);
    float var = ssq[r] * (1.f / 128.f) - mu[r] * mu[r];
    rstd[r] = rsqrtf(var + 1e-5f);
  }
#pragma unroll
  for (int nb = 0; nb < 8; ++nb) {
    float g  = gamma[nb * 16 + lo];
    float bb = beta[nb * 16 + lo];
#pragma unroll
    for (int r = 0; r < 8; ++r) {
      float v = (acc[nb][r] - mu[r]) * rstd[r] * g + bb;
      out[(size_t)(e0 + hi * 8 + r) * DD + nb * 16 + lo] = fmaxf(v, 0.f);
    }
  }
}

// ---------------------------------------------------------------------------
extern "C" void kernel_launch(void* const* d_in, const int* in_sizes, int n_in,
                              void* d_out, int out_size, void* d_ws, size_t ws_size,
                              hipStream_t stream) {
  const float* x       = (const float*)d_in[0];
  const float* h       = (const float*)d_in[1];
  const float* snorm_n = (const float*)d_in[2];
  /* d_in[3] = snorm_e, unused by the reference math */
  const float* W1      = (const float*)d_in[4];
  const float* W2      = (const float*)d_in[5];
  const float* gamma   = (const float*)d_in[6];
  const float* beta    = (const float*)d_in[7];
  const int*   src     = (const int*)d_in[8];
  const int*   dst     = (const int*)d_in[9];
  float*       out     = (float*)d_out;

  char* ws = (char*)d_ws;
  float*    mnode = (float*)(ws + 0);
  uint16_t* mbf   = (uint16_t*)(ws + 20480000);
  uint16_t* xbf   = (uint16_t*)(ws + 30720000);
  uint16_t* w1b   = (uint16_t*)(ws + 40960000);
  uint16_t* w2b   = (uint16_t*)(ws + 41058304);

  // 1) zero mailbox (5.12M floats = 1.28M float4)
  zero_f32_kernel<<<5000, 256, 0, stream>>>((float4*)mnode, 1280000);
  // 2) pack weights into WMMA B-fragment order (tiny, redone every call)
  pack_wB_kernel<<<192, 256, 0, stream>>>(W1, w1b, 12, 384, 49152);
  pack_wB_kernel<<<64, 256, 0, stream>>>(W2, w2b, 4, 128, 16384);
  // 3) x -> bf16 (x rows are gathered ~32x each; convert once)
  cvt_bf16_kernel<<<10000, 256, 0, stream>>>(x, (uint32_t*)xbf, 2560000);
  // 4) GEMM1 + relu + atomic scatter : 40000 waves (16 edges each), 96KB LDS
  gemm1_scatter_kernel<<<5000, 256, 98304, stream>>>(xbf, h, w1b, src, dst, mnode);
  // 5) m_node -> bf16
  cvt_bf16_kernel<<<10000, 256, 0, stream>>>(mnode, (uint32_t*)mbf, 2560000);
  // 6) GEMM2 + snorm + LN + relu, 32KB LDS
  gemm2_ln_kernel<<<5000, 256, 32768, stream>>>(mbf, w2b, src, snorm_n, gamma, beta, out);
}
